// Model_70961449664567
// MI455X (gfx1250) — compile-verified, matrix operand order checked by reading, not run on hardware
//
#include <hip/hip_runtime.h>

// ---------------------------------------------------------------------------
// CDNA5 (gfx1250) implementation. wave32, WMMA bf16 16x16x32 with f32 accum.
// Dominant cost: 3x hop GEMMs [10000x10000]x[10000x158] = 9.5e10 FLOP with
// AI=79 FLOP/B vs the streamed hop operand -> HBM-bound at 23.3 TB/s
// (~55us floor) provided each hop element is read from HBM EXACTLY once.
// So: one block computes the full 160-col output for its rows (no column
// re-streaming), split-K (grid.y) restores block-level parallelism with
// f32 atomic combine. X (3.2 MB bf16) + weights stay L2-resident (192 MB).
// B tiles move via the Tensor Data Mover (tensor_load_to_lds, double-buffered,
// s_wait_tensorcnt-pipelined); A fragments are register-direct global loads;
// B fragments are register-pipelined so ds_load_b128 hides under the WMMAs.
// ---------------------------------------------------------------------------

typedef __bf16 v16bf __attribute__((ext_vector_type(16)));
typedef float  v8f   __attribute__((ext_vector_type(8)));
typedef unsigned int v4u __attribute__((ext_vector_type(4)));
typedef int v8i __attribute__((ext_vector_type(8)));
typedef int v4i __attribute__((ext_vector_type(4)));

#define NN    10000     // nodes
#define NEDGE 320000    // edges
#define DPAD  160       // 158 padded to 5*32
#define H     256
#define KSPLIT 4        // K-range splits for the hop GEMM

#if defined(__gfx1250__) && __has_builtin(__builtin_amdgcn_tensor_load_to_lds)
#define HAVE_TDM 1
#if __has_include(<hip/amd_detail/amd_gfx1250_TDM.h>)
#define TDM_6ARG 1      // therock-10.0 headers -> 6-arg builtin
#else
#define TDM_6ARG 0      // ROCm 7.2 -> 5-arg builtin
#endif
#else
#define HAVE_TDM 0
#endif

static __device__ __forceinline__ int swz_e(int k, int* khalf) {
    // Map K index (0..31) of a bf16 fragment to (lane-half, element)
    // per CDNA5 ISA 7.12.2 "16-bit A-Matrix 16x32".
    int kh = (k >> 3) & 1;
    *khalf = kh;
    return (k < 16) ? (k - kh * 8) : (k - 8 - kh * 8);
}

// ---- A fragment: two contiguous 8-element runs of one row, straight from
// ---- global memory into registers (no LDS round-trip).
template <bool A_F32, bool GUARD>
static __device__ __forceinline__ v16bf load_a_frag(const void* Ap, int lda, int gr,
                                                    int k0, int khalf, int M, int K) {
    union { v16bf v; __bf16 e[16]; unsigned int u[8]; } f;
    if (gr < M) {
        const int kb0 = k0 + khalf * 8;        // elements e=0..7
        const int kb1 = k0 + 16 + khalf * 8;   // elements e=8..15
        if (A_F32) {
            const float* row = (const float*)Ap + (size_t)gr * lda;
            if (!GUARD) {
                const float4 a = *(const float4*)(row + kb0);
                const float4 b = *(const float4*)(row + kb0 + 4);
                const float4 c = *(const float4*)(row + kb1);
                const float4 d = *(const float4*)(row + kb1 + 4);
                f.e[0] = (__bf16)a.x;  f.e[1] = (__bf16)a.y;
                f.e[2] = (__bf16)a.z;  f.e[3] = (__bf16)a.w;
                f.e[4] = (__bf16)b.x;  f.e[5] = (__bf16)b.y;
                f.e[6] = (__bf16)b.z;  f.e[7] = (__bf16)b.w;
                f.e[8] = (__bf16)c.x;  f.e[9] = (__bf16)c.y;
                f.e[10] = (__bf16)c.z; f.e[11] = (__bf16)c.w;
                f.e[12] = (__bf16)d.x; f.e[13] = (__bf16)d.y;
                f.e[14] = (__bf16)d.z; f.e[15] = (__bf16)d.w;
                // stream hint: two K tiles ahead of the hop row
                __builtin_prefetch((const void*)(row + kb0 + 64), 0, 1);
            } else {
#pragma unroll
                for (int j = 0; j < 8; ++j)
                    f.e[j] = (kb0 + j < K) ? (__bf16)row[kb0 + j] : (__bf16)0.0f;
#pragma unroll
                for (int j = 0; j < 8; ++j)
                    f.e[8 + j] = (kb1 + j < K) ? (__bf16)row[kb1 + j] : (__bf16)0.0f;
            }
        } else {
            const __bf16* row = (const __bf16*)Ap + (size_t)gr * lda;
            if (!GUARD) {
                const uint4 lo = *(const uint4*)(row + kb0);
                const uint4 hi = *(const uint4*)(row + kb1);
                f.u[0] = lo.x; f.u[1] = lo.y; f.u[2] = lo.z; f.u[3] = lo.w;
                f.u[4] = hi.x; f.u[5] = hi.y; f.u[6] = hi.z; f.u[7] = hi.w;
            } else {
#pragma unroll
                for (int j = 0; j < 8; ++j)
                    f.e[j] = (kb0 + j < K) ? row[kb0 + j] : (__bf16)0.0f;
#pragma unroll
                for (int j = 0; j < 8; ++j)
                    f.e[8 + j] = (kb1 + j < K) ? row[kb1 + j] : (__bf16)0.0f;
            }
        }
    } else {
#pragma unroll
        for (int j = 0; j < 8; ++j) f.u[j] = 0u;
    }
    return f.v;
}

#if HAVE_TDM
// TDM load of B tile [32 rows x ncols] (bf16) into LDS row-major.
// tensor_dim1 = K - k0 => the TDM zero-fills the K tail automatically.
static __device__ __forceinline__ void tdm_issue_b(const __bf16* Bt, int ldb, int K,
                                                   int k0, int ncols, unsigned lds_off) {
    const unsigned long long ga =
        (unsigned long long)(size_t)(Bt + (size_t)k0 * ldb);
    const int rows = K - k0;
    v4u g0;
    g0.x = 1u;                                              // count=1, user mode
    g0.y = lds_off;                                         // lds_addr
    g0.z = (unsigned)ga;                                    // global_addr lo
    g0.w = (unsigned)((ga >> 32) & 0x01FFFFFFull) | (2u << 30);  // hi | type=2
    v8i g1;
    g1[0] = 1 << 16;                                        // data_size = 2B
    g1[1] = ncols << 16;                                    // tensor_dim0
    g1[2] = (rows & 0xFFFF) << 16;                          // tensor_dim1 lo
    g1[3] = ((rows >> 16) & 0xFFFF) | (ncols << 16);        // dim1 hi | tile_dim0
    g1[4] = 32;                                             // tile_dim1 = 32
    g1[5] = ldb;                                            // dim0 stride (elems)
    g1[6] = 0;
    g1[7] = 0;
    const v4i z4 = {0, 0, 0, 0};
#if TDM_6ARG
    const v8i z8 = {0, 0, 0, 0, 0, 0, 0, 0};
    __builtin_amdgcn_tensor_load_to_lds(g0, g1, z4, z4, z8, 0);
#else
    __builtin_amdgcn_tensor_load_to_lds(g0, g1, z4, z4, 0);
#endif
}
#endif

// ---------------------------------------------------------------------------
// Tiled GEMM: C[M x NCT*32] = A[M x K] * B[K x NCT*32] (+bias)(+C / atomic)
// 256 threads = 8 waves; block tile = 128 rows x FULL output width, so the
// streamed A operand is read exactly once. grid.y = ksplit K-range splits
// (atomic f32 combine when ksplit > 1).
// ---------------------------------------------------------------------------
template <bool A_F32, int NCT>
__global__ __launch_bounds__(256) void k_gemm(const void* __restrict__ Ap, int lda,
                                              const __bf16* __restrict__ B, int ldb,
                                              float* __restrict__ C, int ldc,
                                              const float* __restrict__ bias,
                                              int M, int K, int accum, int ksplit) {
    constexpr int NCOLS = NCT * 32;
    __shared__ __bf16 sB[32 * NCOLS];      // swizzled fragment layout
#if HAVE_TDM
    __shared__ __bf16 rm[2][32 * NCOLS];   // row-major TDM landing buffers
#endif
    const int tid  = threadIdx.x;
    const int wave = tid >> 5;
    const int lane = tid & 31;
    const int row0 = blockIdx.x * 128;

    const int khalf = lane >> 4;
    const int gr    = row0 + wave * 16 + (lane & 15);   // this lane's A row

    const int ntiles   = (K + 31) >> 5;
    const int tilesPer = (ntiles + ksplit - 1) / ksplit;
    const int kt0 = blockIdx.y * tilesPer;
    const int kt1 = (kt0 + tilesPer < ntiles) ? (kt0 + tilesPer) : ntiles;

    // B staging mapping: thread -> (row br, col groups bc0+32*rep)
    const int br  = tid >> 3;
    const int bc0 = (tid & 7) * 4;
    int bkh;
    const int be = swz_e(br, &bkh);

    v8f acc[2 * NCT] = {};

#if HAVE_TDM
    if (wave == 0 && kt0 < kt1)
        tdm_issue_b(B, ldb, K, kt0 << 5, NCOLS, (unsigned)(size_t)&rm[0][0]);
#endif

    for (int i = kt0; i < kt1; ++i) {
        const int k0 = i << 5;
        const bool tail = (k0 + 32 > K);

        // A fragment for this K tile (registers only)
        v16bf a;
        if (!tail) a = load_a_frag<A_F32, false>(Ap, lda, gr, k0, khalf, M, K);
        else       a = load_a_frag<A_F32, true>(Ap, lda, gr, k0, khalf, M, K);

#if HAVE_TDM
        if (wave == 0) __builtin_amdgcn_s_wait_tensorcnt((short)0);
        __syncthreads();   // rm[i&1] ready; prior sB fragment reads retired
        {
            const __bf16* rmb = &rm[i & 1][(size_t)br * NCOLS];
#pragma unroll
            for (int rep = 0; rep < NCT; ++rep) {
                const int c0r = bc0 + 32 * rep;
                const unsigned long long raw =
                    *(const unsigned long long*)&rmb[c0r];
                const __bf16* e = (const __bf16*)&raw;
#pragma unroll
                for (int j = 0; j < 4; ++j) {
                    const int c = c0r + j;
                    sB[((((c >> 4) * 32) + bkh * 16 + (c & 15)) << 4) + be] = e[j];
                }
            }
        }
        if (wave == 0 && i + 1 < kt1)
            tdm_issue_b(B, ldb, K, (i + 1) << 5, NCOLS,
                        (unsigned)(size_t)&rm[(i + 1) & 1][0]);
#else
        __syncthreads();   // prior sB fragment reads retired
        {
            const __bf16* Brow = B + (size_t)(k0 + br) * ldb;
            const bool inb = !tail || (k0 + br) < K;
#pragma unroll
            for (int rep = 0; rep < NCT; ++rep) {
                const int c0r = bc0 + 32 * rep;
                unsigned long long raw = 0;
                if (inb) raw = *(const unsigned long long*)&Brow[c0r];
                const __bf16* e = (const __bf16*)&raw;
#pragma unroll
                for (int j = 0; j < 4; ++j) {
                    const int c = c0r + j;
                    sB[((((c >> 4) * 32) + bkh * 16 + (c & 15)) << 4) + be] = e[j];
                }
            }
        }
#endif
        __syncthreads();   // sB tile ready

        // Register-pipelined B fragments: load nb+1 before the WMMA on nb so
        // the ds_load_b128 pair hides under the matrix op (avoids a full
        // s_wait_dscnt 0x0 in front of every WMMA).
        v16bf bcur = *(const v16bf*)&sB[lane << 4];
#pragma unroll
        for (int nb = 0; nb < 2 * NCT; ++nb) {
            v16bf bnext = bcur;
            if (nb + 1 < 2 * NCT)
                bnext = *(const v16bf*)&sB[((nb + 1) * 32 + lane) << 4];
            acc[nb] = __builtin_amdgcn_wmma_f32_16x16x32_bf16(
                false, a, false, bcur, (short)0, acc[nb], false, false);
            bcur = bnext;
        }
    }

    // ---- epilogue: C/D layout (ISA 7.12.2): col = lane%16, row r + 8*(lane/16)
    const int mrow_off = 8 * (lane >> 4);
    const int ncol     = lane & 15;
#pragma unroll
    for (int r = 0; r < 8; ++r) {
        const int m = row0 + wave * 16 + mrow_off + r;
        if (m < M) {
#pragma unroll
            for (int nb = 0; nb < 2 * NCT; ++nb) {
                const int col = nb * 16 + ncol;
                float v = acc[nb][r];
                const size_t idx = (size_t)m * ldc + col;
                if (ksplit > 1) {
                    atomicAdd(&C[idx], v);      // split-K combine
                } else {
                    if (bias)  v += bias[col];
                    if (accum) v += C[idx];
                    C[idx] = v;
                }
            }
        }
    }
}

// ---------------------------------------------------------------------------
// Elementwise / graph kernels
// ---------------------------------------------------------------------------
__global__ void k_fill(float* p, float v, int n) {
    const int i = blockIdx.x * blockDim.x + threadIdx.x;
    if (i < n) p[i] = v;
}

__global__ void k_deg(const long long* __restrict__ edges, float* __restrict__ deg, int E) {
    const int e = blockIdx.x * blockDim.x + threadIdx.x;
    if (e < E) atomicAdd(&deg[(int)edges[E + e]], 1.0f);
}

__global__ void k_dinv(float* d, int n) {
    const int i = blockIdx.x * blockDim.x + threadIdx.x;
    if (i < n) { const float v = d[i]; d[i] = (v > 0.0f) ? rsqrtf(v) : 0.0f; }
}

// X = concat(x, walk_feats) padded to 160 cols, bf16
__global__ void k_build_xbf(const float* __restrict__ x, const float* __restrict__ walk,
                            __bf16* __restrict__ xbf, int n) {
    const int i = blockIdx.x * blockDim.x + threadIdx.x;
    if (i < n * DPAD) {
        const int node = i / DPAD, c = i % DPAD;
        float v = 0.0f;
        if (c < 128)      v = x[node * 128 + c];
        else if (c < 158) v = walk[node * 30 + (c - 128)];
        xbf[i] = (__bf16)v;
    }
}

__global__ void k_cvt_bf16(const float* __restrict__ s, __bf16* __restrict__ d, int n) {
    const int i = blockIdx.x * blockDim.x + threadIdx.x;
    if (i < n) d[i] = (__bf16)s[i];
}

// f32 [srcRows x cols] -> bf16 [dstRows x cols] with zero row padding
__global__ void k_cvt_pad(const float* __restrict__ s, __bf16* __restrict__ d,
                          int srcRows, int cols, int dstRows) {
    const int i = blockIdx.x * blockDim.x + threadIdx.x;
    if (i < dstRows * cols) {
        const int r = i / cols;
        d[i] = (__bf16)((r < srcRows) ? s[i] : 0.0f);
    }
}

// per-edge scatter: acc[dst] += hw[src] * dinv[src]*dinv[dst]  (4 feats/thread)
__global__ void k_scatter(const long long* __restrict__ edges, const float* __restrict__ dinv,
                          const float* __restrict__ hw, float* __restrict__ acc, int E) {
    const int i = blockIdx.x * blockDim.x + threadIdx.x;
    const int e = i >> 6;
    if (e < E) {
        const int fq  = (i & 63) << 2;
        const int src = (int)edges[e];
        const int dst = (int)edges[E + e];
        const float nrm = dinv[src] * dinv[dst];
        const float4 m = *(const float4*)&hw[(size_t)src * H + fq];
        float* d = &acc[(size_t)dst * H + fq];
        atomicAdd(d + 0, m.x * nrm);
        atomicAdd(d + 1, m.y * nrm);
        atomicAdd(d + 2, m.z * nrm);
        atomicAdd(d + 3, m.w * nrm);
    }
}

// h = relu(acc + hw * dinv^2 + b); optionally emit bf16 copy and/or d_out slice
__global__ void k_gcn_finish(const float* __restrict__ acc, const float* __restrict__ hw,
                             const float* __restrict__ dinv, const float* __restrict__ b,
                             __bf16* __restrict__ hbf, float* __restrict__ outcol, int n) {
    const int i = blockIdx.x * blockDim.x + threadIdx.x;
    if (i < n * H) {
        const int node = i >> 8, f = i & 255;
        const float dv = dinv[node];
        float v = acc[i] + hw[i] * dv * dv + b[f];
        v = v > 0.0f ? v : 0.0f;
        if (hbf)    hbf[i] = (__bf16)v;
        if (outcol) outcol[(size_t)node * 512 + f] = v;
    }
}

// column-wise sum / sum-of-squares (training-mode BatchNorm)
__global__ void k_bn_stats(const float* __restrict__ Hm, float* __restrict__ sum,
                           float* __restrict__ sumsq, int n) {
    const int f  = threadIdx.x;       // 256 columns
    const int r0 = blockIdx.x * 128;
    float s = 0.0f, s2 = 0.0f;
    for (int r = 0; r < 128; ++r) {
        const int node = r0 + r;
        if (node < n) {
            const float v = Hm[(size_t)node * H + f];
            s += v; s2 += v * v;
        }
    }
    atomicAdd(&sum[f], s);
    atomicAdd(&sumsq[f], s2);
}

// bf16( relu( g*(h-mu)*rsqrt(var+eps) + be ) )   (b1 cancels inside BN)
__global__ void k_bn_apply(const float* __restrict__ Hm, const float* __restrict__ sum,
                           const float* __restrict__ sumsq, const float* __restrict__ g,
                           const float* __restrict__ be, __bf16* __restrict__ Hbf, int n) {
    const int i = blockIdx.x * blockDim.x + threadIdx.x;
    if (i < n * H) {
        const int f = i & 255;
        const float inv_n = 1.0f / (float)n;
        const float mu  = sum[f] * inv_n;
        const float var = sumsq[f] * inv_n - mu * mu;
        float v = g[f] * (Hm[i] - mu) * rsqrtf(var + 1e-5f) + be[f];
        v = v > 0.0f ? v : 0.0f;
        Hbf[i] = (__bf16)v;
    }
}

// ---------------------------------------------------------------------------
static inline int cdiv(int a, int b) { return (a + b - 1) / b; }

extern "C" void kernel_launch(void* const* d_in, const int* in_sizes, int n_in,
                              void* d_out, int out_size, void* d_ws, size_t ws_size,
                              hipStream_t stream) {
    (void)in_sizes; (void)n_in; (void)out_size; (void)ws_size;

    const float*     x     = (const float*)d_in[0];
    const long long* edges = (const long long*)d_in[1];
    const float*     walk  = (const float*)d_in[2];
    const float*     hops[3] = {(const float*)d_in[3], (const float*)d_in[4],
                                (const float*)d_in[5]};
    const float* gW[3] = {(const float*)d_in[6], (const float*)d_in[8], (const float*)d_in[10]};
    const float* gb[3] = {(const float*)d_in[7], (const float*)d_in[9], (const float*)d_in[11]};
    const float* sW1 = (const float*)d_in[12];   // [4,158,256]
    // d_in[13] = sub_b1: cancelled by BatchNorm mean subtraction
    const float* sg  = (const float*)d_in[14];   // [4,256]
    const float* sbe = (const float*)d_in[15];   // [4,256]
    const float* sW2 = (const float*)d_in[16];   // [4,256,256]
    const float* sb2 = (const float*)d_in[17];   // [4,256]
    float* out = (float*)d_out;                  // [N, 512]

    // ---- workspace carve-up ----
    char* w = (char*)d_ws;
    size_t off = 0;
    auto wsAlloc = [&](size_t bytes) -> void* {
        void* p = w + off;
        off = (off + bytes + 255) & ~(size_t)255;
        return p;
    };
    float*  dinv  = (float*)wsAlloc((size_t)NN * 4);
    __bf16* Xbf   = (__bf16*)wsAlloc((size_t)NN * DPAD * 2);
    float*  AX    = (float*)wsAlloc((size_t)NN * DPAD * 4);
    __bf16* AXbf  = (__bf16*)wsAlloc((size_t)NN * DPAD * 2);
    float*  hw    = (float*)wsAlloc((size_t)NN * H * 4);
    float*  hacc  = (float*)wsAlloc((size_t)NN * H * 4);
    __bf16* hbf   = (__bf16*)wsAlloc((size_t)NN * H * 2);
    float*  H1    = (float*)wsAlloc((size_t)NN * H * 4);
    __bf16* Hbf   = (__bf16*)wsAlloc((size_t)NN * H * 2);
    __bf16* Wbf   = (__bf16*)wsAlloc((size_t)H * H * 2);
    float*  bnsum = (float*)wsAlloc(H * 4);
    float*  bnsq  = (float*)wsAlloc(H * 4);

    const dim3 blk(256);
    const int rowBlocks = cdiv(NN, 128);   // 79

    // ---- degrees / symmetric norm (self loops folded in as init=1) ----
    k_fill<<<cdiv(NN, 256), blk, 0, stream>>>(dinv, 1.0f, NN);
    k_deg<<<cdiv(NEDGE, 256), blk, 0, stream>>>(edges, dinv, NEDGE);
    k_dinv<<<cdiv(NN, 256), blk, 0, stream>>>(dinv, NN);

    // ---- X = concat(x, walk) -> bf16, padded to 160; stays L2-resident ----
    k_build_xbf<<<cdiv(NN * DPAD, 256), blk, 0, stream>>>(x, walk, Xbf, NN);

    // ---- 3-layer GCN ----
    for (int l = 0; l < 3; ++l) {
        const int Kl = (l == 0) ? 128 : H;
        k_cvt_bf16<<<cdiv(Kl * H, 256), blk, 0, stream>>>(gW[l], Wbf, Kl * H);
        const __bf16* Abf = (l == 0) ? Xbf : hbf;
        const int lda     = (l == 0) ? DPAD : H;
        k_gemm<false, 8><<<dim3(rowBlocks, 1), blk, 0, stream>>>(
            Abf, lda, Wbf, H, hw, H, nullptr, NN, Kl, 0, 1);
        k_fill<<<cdiv(NN * H, 256), blk, 0, stream>>>(hacc, 0.0f, NN * H);
        k_scatter<<<cdiv(NEDGE * 64, 256), blk, 0, stream>>>(edges, dinv, hw, hacc, NEDGE);
        if (l < 2)
            k_gcn_finish<<<cdiv(NN * H, 256), blk, 0, stream>>>(hacc, hw, dinv, gb[l],
                                                                hbf, nullptr, NN);
        else
            k_gcn_finish<<<cdiv(NN * H, 256), blk, 0, stream>>>(hacc, hw, dinv, gb[l],
                                                                nullptr, out, NN);
    }

    // ---- subgraph branch: s=0 uses X; s>=1 uses hop_s @ X (the big GEMMs) ----
    for (int s = 0; s < 4; ++s) {
        const __bf16* Abf;
        if (s == 0) {
            Abf = Xbf;
        } else {
            // HBM-bound streamed GEMM: hop read exactly once; split-K combine.
            k_fill<<<cdiv(NN * DPAD, 256), blk, 0, stream>>>(AX, 0.0f, NN * DPAD);
            k_gemm<true, 5><<<dim3(rowBlocks, KSPLIT), blk, 0, stream>>>(
                hops[s - 1], NN, Xbf, DPAD, AX, DPAD, nullptr, NN, NN, 0, KSPLIT);
            k_cvt_bf16<<<cdiv(NN * DPAD, 256), blk, 0, stream>>>(AX, AXbf, NN * DPAD);
            Abf = AXbf;
        }
        // Linear1 (b1 dropped: cancels in BN), K padded 158->160
        k_cvt_pad<<<cdiv(DPAD * H, 256), blk, 0, stream>>>(sW1 + (size_t)s * 158 * H,
                                                           Wbf, 158, H, DPAD);
        k_gemm<false, 8><<<dim3(rowBlocks, 1), blk, 0, stream>>>(
            Abf, DPAD, Wbf, H, H1, H, nullptr, NN, DPAD, 0, 1);
        // BatchNorm (training-mode batch stats) + ReLU -> bf16
        k_fill<<<1, blk, 0, stream>>>(bnsum, 0.0f, H);
        k_fill<<<1, blk, 0, stream>>>(bnsq, 0.0f, H);
        k_bn_stats<<<cdiv(NN, 128), blk, 0, stream>>>(H1, bnsum, bnsq, NN);
        k_bn_apply<<<cdiv(NN * H, 256), blk, 0, stream>>>(H1, bnsum, bnsq,
                                                          sg + s * H, sbe + s * H,
                                                          Hbf, NN);
        // Linear2, accumulate straight into out[:, 256:512] with b2 fused
        k_cvt_bf16<<<cdiv(H * H, 256), blk, 0, stream>>>(sW2 + (size_t)s * H * H,
                                                         Wbf, H * H);
        k_gemm<false, 8><<<dim3(rowBlocks, 1), blk, 0, stream>>>(
            Hbf, H, Wbf, H, out + 256, 512, sb2 + s * H, NN, H, (s > 0) ? 1 : 0, 1);
    }
}
